// MultiHeadBatteryAttention_35184372089391
// MI455X (gfx1250) — compile-verified, matrix operand order checked by reading, not run on hardware
//
#include <hip/hip_runtime.h>

// ---------------------------------------------------------------------------
// MultiHeadBatteryAttention for MI455X (gfx1250, wave32, WMMA bf16)
// B=2, S=2048, D=1024, H=16, DK=64
// ---------------------------------------------------------------------------

typedef __bf16 bf16;
typedef __attribute__((ext_vector_type(16))) __bf16 v16bf;
typedef __attribute__((ext_vector_type(8)))  __bf16 v8bf;
typedef __attribute__((ext_vector_type(4)))  __bf16 v4bf;
typedef __attribute__((ext_vector_type(8)))  float  v8f;

#define BATCH 2
#define SEQ   2048
#define DMODEL 1024
#define HEADS 16
#define DK    64

// ---------------------------------------------------------------------------
// WMMA bf16 fragment loader (A 16x32 and B 32x16 share the layout when the
// B source is stored [n][k]).  lane = idx + 16*kg; element j holds
// k = (j/8)*16 + kg*8 + (j%8): two contiguous 16B chunks per lane.
// ---------------------------------------------------------------------------
template <typename P>
__device__ __forceinline__ v16bf load_frag(const P* basep, int stride,
                                           int row0, int col0, int lane) {
  const bf16* base = (const bf16*)basep;
  int r  = row0 + (lane & 15);
  int kg = (lane >> 4) & 1;
  const bf16* p = base + (size_t)r * stride + col0 + kg * 8;
  v8bf lo = *(const v8bf*)(p);
  v8bf hi = *(const v8bf*)(p + 16);
  return __builtin_shufflevector(lo, hi, 0,1,2,3,4,5,6,7,8,9,10,11,12,13,14,15);
}

__device__ __forceinline__ v8f wmma_bf16(v16bf a, v16bf b, v8f c) {
  return __builtin_amdgcn_wmma_f32_16x16x32_bf16(false, a, false, b,
                                                 (short)0, c, false, false);
}

// ---------------------------------------------------------------------------
// Kernel 1: fused QKV projection (fp32 in -> bf16 out).
// M=B*S=4096, N=D=1024, K=D=1024.  blockIdx.z selects {Q,K,V}.
// Q,K written [B,H,S,DK]; V written transposed [B,H,DK,S].
// Tile 128(M) x 64(N), 8 waves of 32x32.  Within a block: b=bm>>4, h=bn are
// uniform, so the epilogue is fully branchless with two uniform strides.
// ---------------------------------------------------------------------------
__global__ __launch_bounds__(256)
void proj_qkv_kernel(const float* __restrict__ q_in,
                     const float* __restrict__ k_in,
                     const float* __restrict__ v_in,
                     const float* __restrict__ Wq,
                     const float* __restrict__ Wk,
                     const float* __restrict__ Wv,
                     bf16* __restrict__ Qp,
                     bf16* __restrict__ Kp,
                     bf16* __restrict__ Vt) {
  __shared__ bf16 sA[128 * 32];  // [m][k]
  __shared__ bf16 sB[64 * 32];   // [n][k]  (W transposed on the fly)

  const int mat = blockIdx.z;
  const float* X = (mat == 0) ? q_in : (mat == 1) ? k_in : v_in;
  const float* W = (mat == 0) ? Wq   : (mat == 1) ? Wk   : Wv;

  const int bn = blockIdx.x;    // N/64 = 16  (== head index)
  const int bm = blockIdx.y;    // M/128 = 32
  const int tid = threadIdx.x;
  const int lane = tid & 31;
  const int wid = tid >> 5;
  const int wm = wid >> 1;      // 0..3
  const int wn = wid & 1;       // 0..1

  v8f acc[2][2] = {};

  for (int kk = 0; kk < DMODEL / 32; ++kk) {
    if (kk + 1 < DMODEL / 32) {
      // next A tile: 128 rows x 32 f32; 256 lanes cover 128 rows x 2 x 64B
      __builtin_prefetch(X + (size_t)(bm * 128 + (tid >> 1)) * DMODEL
                           + (kk + 1) * 32 + (tid & 1) * 16, 0, 1);
      // next B tile: 32 rows x 64 f32; 256 lanes cover 32 rows x 8 x 32B
      __builtin_prefetch(W + (size_t)((kk + 1) * 32 + (tid >> 3)) * DMODEL
                           + bn * 64 + (tid & 7) * 8, 0, 1);
    }
    // --- A tile: 128x32 fp32 -> bf16, 1024 float4, 4 per thread
    for (int it = 0; it < 4; ++it) {
      int linear = tid + 256 * it;
      int row = linear >> 3;
      int c4  = linear & 7;
      const float4 f = *(const float4*)(X + (size_t)(bm * 128 + row) * DMODEL
                                          + kk * 32 + c4 * 4);
      v4bf o; o[0] = (bf16)f.x; o[1] = (bf16)f.y; o[2] = (bf16)f.z; o[3] = (bf16)f.w;
      *(v4bf*)(&sA[row * 32 + c4 * 4]) = o;
    }
    // --- B tile transposed: W[k][n] -> sB[n][k], 512 float4, 2 per thread
    for (int it = 0; it < 2; ++it) {
      int linear = tid + 256 * it;
      int krow = linear >> 4;
      int c4   = linear & 15;
      const float4 f = *(const float4*)(W + (size_t)(kk * 32 + krow) * DMODEL
                                          + bn * 64 + c4 * 4);
      sB[(c4 * 4 + 0) * 32 + krow] = (bf16)f.x;
      sB[(c4 * 4 + 1) * 32 + krow] = (bf16)f.y;
      sB[(c4 * 4 + 2) * 32 + krow] = (bf16)f.z;
      sB[(c4 * 4 + 3) * 32 + krow] = (bf16)f.w;
    }
    __syncthreads();

    v16bf af[2], bfr[2];
    af[0]  = load_frag(sA, 32, wm * 32 + 0,  0, lane);
    af[1]  = load_frag(sA, 32, wm * 32 + 16, 0, lane);
    bfr[0] = load_frag(sB, 32, wn * 32 + 0,  0, lane);
    bfr[1] = load_frag(sB, 32, wn * 32 + 16, 0, lane);
    for (int mi = 0; mi < 2; ++mi)
      for (int ni = 0; ni < 2; ++ni)
        acc[mi][ni] = wmma_bf16(af[mi], bfr[ni], acc[mi][ni]);
    __syncthreads();
  }

  // --- branchless epilogue --------------------------------------------------
  const int mh = (lane >> 4) * 8;
  const int nl = lane & 15;
  const int b = bm >> 4;                 // batch (uniform per block)
  const int h = bn;                      // head  (uniform per block)
  bf16* dstp = (mat == 0) ? Qp : (mat == 1) ? Kp : Vt;
  bf16* dbase = dstp + (size_t)(b * HEADS + h) * (SEQ * DK);
  const int Cs = (mat == 2) ? 1 : DK;    // stride of s
  const int Cd = (mat == 2) ? SEQ : 1;   // stride of dk
  const int s0 = bm * 128 - b * SEQ + wm * 32 + mh;
  const int d0 = wn * 32 + nl;
  for (int mi = 0; mi < 2; ++mi)
    for (int ni = 0; ni < 2; ++ni)
      for (int r = 0; r < 8; ++r)
        dbase[(size_t)(s0 + mi * 16 + r) * Cs + (size_t)(d0 + ni * 16) * Cd] =
            (bf16)acc[mi][ni][r];
}

// ---------------------------------------------------------------------------
// Kernel 2: fused scores + softmax + attn-write + context.
// One workgroup = 16 query rows of one (b,h); full 16x2048 score strip in
// LDS (128KB f32 + 64KB bf16 + 4KB reduce < 320KB/WGP).
// ---------------------------------------------------------------------------
__global__ __launch_bounds__(256)
void attn_kernel(const bf16* __restrict__ Qp,
                 const bf16* __restrict__ Kp,
                 const bf16* __restrict__ Vt,
                 const int* __restrict__ mask,
                 const float* __restrict__ sensor_w,
                 const float* __restrict__ sensor_attn,
                 const float* __restrict__ temporal_bias,
                 float* __restrict__ attn_out,
                 bf16* __restrict__ Ctx) {
  extern __shared__ char smem[];
  float* sc   = (float*)smem;                                   // [16][2048]
  bf16*  pr   = (bf16*)(smem + 16 * SEQ * 4);                   // [16][2048]
  float* cacc = (float*)(smem + 16 * SEQ * 4 + 16 * SEQ * 2);   // [16][64]
  float* rbias = cacc + 16 * DK;                                // [16]

  const int bh = blockIdx.y;             // 0..31
  const int b = bh >> 4, h = bh & 15;
  const int q0 = blockIdx.x * 16;
  const int tid = threadIdx.x;
  const int lane = tid & 31;
  const int wid = tid >> 5;

  for (int i = tid; i < 16 * DK; i += 256) cacc[i] = 0.f;
  if (tid < 16) {
    float sb = 0.f;
    for (int f = 0; f < 4; ++f)
      sb += sensor_attn[h * 4 + f] *
            sensor_w[((size_t)b * SEQ + q0 + tid) * 4 + f];
    rbias[tid] = sb + temporal_bias[h];
  }

  // --- persistent Q fragments
  const bf16* Qb = Qp + ((size_t)bh * SEQ + q0) * DK;
  const v16bf qf0 = load_frag(Qb, DK, 0, 0,  lane);
  const v16bf qf1 = load_frag(Qb, DK, 0, 32, lane);

  // --- scores: wave owns key tiles kb, kb+8, ...
  const bf16* Kb = Kp + (size_t)bh * SEQ * DK;
  const int mh = (lane >> 4) * 8;
  const int nl = lane & 15;
  for (int kb = wid; kb < SEQ / 16; kb += 8) {
    if (kb + 8 < SEQ / 16)   // prefetch next K tile: 16 rows x 128B = 32x64B
      __builtin_prefetch(Kb + (size_t)(kb + 8) * 16 * DK + lane * 32, 0, 1);
    v16bf kf0 = load_frag(Kb, DK, kb * 16, 0,  lane);
    v16bf kf1 = load_frag(Kb, DK, kb * 16, 32, lane);
    v8f a = {};
    a = wmma_bf16(qf0, kf0, a);
    a = wmma_bf16(qf1, kf1, a);
    for (int r = 0; r < 8; ++r)
      sc[(size_t)(mh + r) * SEQ + kb * 16 + nl] = a[r] * 0.125f;  // /sqrt(64)
  }
  __syncthreads();

  // --- softmax: wave w handles rows w and w+8; float4 vectorized
  for (int mi = 0; mi < 2; ++mi) {
    const int m = wid + mi * 8;
    float* row = sc + (size_t)m * SEQ;
    const int* mrow = mask + ((size_t)b * SEQ + q0 + m) * SEQ;
    const float rb = rbias[m];
    float mx = -3.4e38f;
    for (int j = lane; j < SEQ / 4; j += 32) {
      float4 v = *(float4*)(row + 4 * j);
      const int4 mk = *(const int4*)(mrow + 4 * j);
      v.x = (mk.x == 0) ? -1e9f : v.x + rb;
      v.y = (mk.y == 0) ? -1e9f : v.y + rb;
      v.z = (mk.z == 0) ? -1e9f : v.z + rb;
      v.w = (mk.w == 0) ? -1e9f : v.w + rb;
      *(float4*)(row + 4 * j) = v;
      mx = fmaxf(mx, fmaxf(fmaxf(v.x, v.y), fmaxf(v.z, v.w)));
    }
    for (int o = 16; o > 0; o >>= 1) mx = fmaxf(mx, __shfl_xor(mx, o, 32));
    float sum = 0.f;
    for (int j = lane; j < SEQ / 4; j += 32) {
      float4 v = *(float4*)(row + 4 * j);
      v.x = __expf(v.x - mx); v.y = __expf(v.y - mx);
      v.z = __expf(v.z - mx); v.w = __expf(v.w - mx);
      *(float4*)(row + 4 * j) = v;
      sum += (v.x + v.y) + (v.z + v.w);
    }
    for (int o = 16; o > 0; o >>= 1) sum += __shfl_xor(sum, o, 32);
    const float inv = 1.f / sum;
    float* aout = attn_out + ((size_t)bh * SEQ + q0 + m) * SEQ;
    bf16* prow = pr + (size_t)m * SEQ;
    for (int j = lane; j < SEQ / 4; j += 32) {
      float4 v = *(float4*)(row + 4 * j);
      v.x *= inv; v.y *= inv; v.z *= inv; v.w *= inv;
      *(float4*)(aout + 4 * j) = v;      // b128 stores on the 537MB stream
      v4bf pb; pb[0] = (bf16)v.x; pb[1] = (bf16)v.y;
      pb[2] = (bf16)v.z; pb[3] = (bf16)v.w;
      *(v4bf*)(prow + 4 * j) = pb;
    }
  }
  __syncthreads();

  // --- context: (16 x 2048) x (2048 x 64); K split across waves,
  // fp32 partials reduced with ds_add_f32.
  const bf16* Vb = Vt + (size_t)bh * DK * SEQ;
  v8f c0 = {}, c1 = {}, c2 = {}, c3 = {};
  for (int kc = wid; kc < SEQ / 32; kc += 8) {
    v16bf pf = load_frag(pr, SEQ, 0, kc * 32, lane);
    v16bf v0 = load_frag(Vb, SEQ, 0,  kc * 32, lane);
    v16bf v1 = load_frag(Vb, SEQ, 16, kc * 32, lane);
    v16bf v2 = load_frag(Vb, SEQ, 32, kc * 32, lane);
    v16bf v3 = load_frag(Vb, SEQ, 48, kc * 32, lane);
    c0 = wmma_bf16(pf, v0, c0);
    c1 = wmma_bf16(pf, v1, c1);
    c2 = wmma_bf16(pf, v2, c2);
    c3 = wmma_bf16(pf, v3, c3);
  }
  for (int r = 0; r < 8; ++r) {
    atomicAdd(&cacc[(mh + r) * DK + 0  + nl], c0[r]);
    atomicAdd(&cacc[(mh + r) * DK + 16 + nl], c1[r]);
    atomicAdd(&cacc[(mh + r) * DK + 32 + nl], c2[r]);
    atomicAdd(&cacc[(mh + r) * DK + 48 + nl], c3[r]);
  }
  __syncthreads();

  // --- write context bf16 in [b, s, h*64+dk] layout for the output GEMM
  bf16* cbase = Ctx + ((size_t)b * SEQ + q0) * DMODEL + h * DK;
  for (int i = tid; i < 16 * DK; i += 256) {
    int m = i / DK, n = i % DK;
    cbase[(size_t)m * DMODEL + n] = (bf16)cacc[i];
  }
}

// ---------------------------------------------------------------------------
// Kernel 3: output projection.  out = Ctx(bf16) @ Wo(fp32->bf16) + bo, fp32.
// A fragments load straight from global (Ctx already bf16, contiguous 16B
// chunks); only Wo goes through LDS for the transpose+convert.
// ---------------------------------------------------------------------------
__global__ __launch_bounds__(256)
void out_proj_kernel(const bf16* __restrict__ Ctx,
                     const float* __restrict__ Wo,
                     const float* __restrict__ bo,
                     float* __restrict__ out) {
  __shared__ bf16 sB[64 * 32];

  const int bn = blockIdx.x;    // 16
  const int bm = blockIdx.y;    // 32
  const int tid = threadIdx.x;
  const int lane = tid & 31;
  const int wid = tid >> 5;
  const int wm = wid >> 1;
  const int wn = wid & 1;

  v8f acc[2][2] = {};
  const bf16* Abase = Ctx + (size_t)(bm * 128 + wm * 32) * DMODEL;

  for (int kk = 0; kk < DMODEL / 32; ++kk) {
    if (kk + 1 < DMODEL / 32)
      __builtin_prefetch(Wo + (size_t)((kk + 1) * 32 + (tid >> 3)) * DMODEL
                            + bn * 64 + (tid & 7) * 8, 0, 1);
    for (int it = 0; it < 2; ++it) {
      int linear = tid + 256 * it;
      int krow = linear >> 4;
      int c4   = linear & 15;
      const float4 f = *(const float4*)(Wo + (size_t)(kk * 32 + krow) * DMODEL
                                           + bn * 64 + c4 * 4);
      sB[(c4 * 4 + 0) * 32 + krow] = (bf16)f.x;
      sB[(c4 * 4 + 1) * 32 + krow] = (bf16)f.y;
      sB[(c4 * 4 + 2) * 32 + krow] = (bf16)f.z;
      sB[(c4 * 4 + 3) * 32 + krow] = (bf16)f.w;
    }
    __syncthreads();

    v16bf af[2], bfr[2];
    af[0]  = load_frag(Abase, DMODEL, 0,  kk * 32, lane);
    af[1]  = load_frag(Abase, DMODEL, 16, kk * 32, lane);
    bfr[0] = load_frag(sB, 32, wn * 32 + 0,  0, lane);
    bfr[1] = load_frag(sB, 32, wn * 32 + 16, 0, lane);
    for (int mi = 0; mi < 2; ++mi)
      for (int ni = 0; ni < 2; ++ni)
        acc[mi][ni] = wmma_bf16(af[mi], bfr[ni], acc[mi][ni]);
    __syncthreads();
  }

  const int mh = (lane >> 4) * 8;
  const int nl = lane & 15;
  for (int mi = 0; mi < 2; ++mi)
    for (int ni = 0; ni < 2; ++ni)
      for (int r = 0; r < 8; ++r) {
        int m = bm * 128 + wm * 32 + mi * 16 + mh + r;
        int n = bn * 64 + wn * 32 + ni * 16 + nl;
        out[(size_t)m * DMODEL + n] = acc[mi][ni][r] + bo[n];
      }
}

// ---------------------------------------------------------------------------
extern "C" void kernel_launch(void* const* d_in, const int* in_sizes, int n_in,
                              void* d_out, int out_size, void* d_ws, size_t ws_size,
                              hipStream_t stream) {
  const float* query    = (const float*)d_in[0];
  const float* key      = (const float*)d_in[1];
  const float* value    = (const float*)d_in[2];
  const int*   mask     = (const int*)  d_in[3];
  const float* sensor_w = (const float*)d_in[4];
  const float* Wq       = (const float*)d_in[5];
  const float* Wk       = (const float*)d_in[6];
  const float* Wv       = (const float*)d_in[7];
  const float* Wo       = (const float*)d_in[8];
  const float* bo       = (const float*)d_in[9];
  const float* sens_att = (const float*)d_in[10];
  const float* temp_bias= (const float*)d_in[11];

  float* out      = (float*)d_out;                       // [B,S,D]
  float* attn_out = out + (size_t)BATCH * SEQ * DMODEL;  // [B,H,S,S]

  const size_t QKV = (size_t)BATCH * HEADS * SEQ * DK;   // 4,194,304 elems
  bf16* Qp  = (bf16*)d_ws;
  bf16* Kp  = Qp + QKV;
  bf16* Vt  = Kp + QKV;
  bf16* Ctx = Vt + QKV;                                   // [B,S,D] bf16

  proj_qkv_kernel<<<dim3(DMODEL / 64, (BATCH * SEQ) / 128, 3), 256, 0, stream>>>(
      query, key, value, Wq, Wk, Wv, Qp, Kp, Vt);

  const size_t smem = (size_t)16 * SEQ * 4
                    + (size_t)16 * SEQ * 2
                    + (size_t)16 * DK * 4
                    + 16 * 4;
  attn_kernel<<<dim3(SEQ / 16, BATCH * HEADS), 256, smem, stream>>>(
      Qp, Kp, Vt, mask, sensor_w, sens_att, temp_bias, attn_out, Ctx);

  out_proj_kernel<<<dim3(DMODEL / 64, (BATCH * SEQ) / 128), 256, 0, stream>>>(
      Ctx, Wo, bo, out);
}
